// MoE_18751827214915
// MI455X (gfx1250) — matrix operand order repaired and checked by reading, not compiled, not run on hardware
//
#include <hip/hip_runtime.h>

typedef __attribute__((ext_vector_type(16))) __bf16 v16bf;
typedef __attribute__((ext_vector_type(8)))  float  v8f;

#define B_T 4096
#define D_T 1536
#define E_T 64
#define H_T 768
#define C_T 191
#define KT1 48   // D_T / 32 (K tiles, layer 1)
#define NT1 48   // H_T / 16 (N tiles, layer 1)
#define KT2 24   // H_T / 32 (K tiles, layer 2)
#define NT2 12   // 192 / 16 (N tiles, layer 2; C padded 191->192)
#define MT  32   // tokens per MLP block (two 16-row WMMA M tiles)
#define TOPK 8

// K-permutation inside a 32-element K tile so each lane's 16 A-fragment halves
// are contiguous: storage position p holds K-offset perm(p), with
// perm = [0..7, 16..23, 8..15, 24..31]  (an involution).
__device__ __host__ inline int kperm(int p) {
  return (p < 8) ? p : (p < 16) ? p + 8 : (p < 24) ? p - 8 : p;
}

__device__ inline v16bf load_v16(const __bf16* p) { return *(const v16bf*)p; }

// Hint: interleave 1 WMMA with 2 VMEM reads, 6x (one GEMM half-step).
__device__ inline void moe_sched_hint() {
  #pragma unroll
  for (int i = 0; i < 6; ++i) {
    __builtin_amdgcn_sched_group_barrier(0x008, 1, 0);  // 1 MFMA/WMMA
    __builtin_amdgcn_sched_group_barrier(0x020, 2, 0);  // 2 VMEM reads
  }
}

// ---------------------------------------------------------------------------
// Zero-init outputs and accumulators (d_out/d_ws are poisoned by harness).
// ---------------------------------------------------------------------------
__global__ void moe_zero_kernel(float* __restrict__ out, int* __restrict__ counts,
                                float* __restrict__ colsum, float* __restrict__ zsum) {
  long i = (long)blockIdx.x * 256 + threadIdx.x;
  if (i < (long)B_T * C_T) out[i] = 0.0f;
  if (i < E_T) { counts[i] = 0; colsum[i] = 0.0f; }
  if (i == 0) zsum[0] = 0.0f;
}

// ---------------------------------------------------------------------------
// fp32 -> bf16 activation copy with per-32 K-tile permutation (A-fragment
// friendly: one contiguous 32B load per lane per K step).
// ---------------------------------------------------------------------------
__global__ void moe_pack_x_kernel(const float* __restrict__ x, __bf16* __restrict__ xbf) {
  long i = (long)blockIdx.x * 256 + threadIdx.x;
  if (i >= (long)B_T * D_T) return;
  long row = i / D_T;
  int  pos = (int)(i % D_T);
  int  kt  = pos >> 5;
  int  g   = pos & 31;
  xbf[i] = (__bf16)x[row * D_T + kt * 32 + kperm(g)];
}

// ---------------------------------------------------------------------------
// Pack W1 into WMMA B-fragment layout:
//   w1p[e][kt][nt][lane][h] holds W1[e][K][N] with
//   K = kt*32 + kperm((lane>>4)*16 + h),  N = nt*16 + (lane&15)
// ---------------------------------------------------------------------------
__global__ void moe_pack_w1_kernel(const float* __restrict__ W1, __bf16* __restrict__ w1p) {
  long idx = (long)blockIdx.x * 256 + threadIdx.x;
  if (idx >= (long)E_T * KT1 * NT1 * 512) return;
  int h    = (int)(idx & 15);
  int lane = (int)((idx >> 4) & 31);
  long rest = idx >> 9;
  int nt = (int)(rest % NT1); rest /= NT1;
  int kt = (int)(rest % KT1);
  int e  = (int)(rest / KT1);
  int K = kt * 32 + kperm(((lane >> 4) << 4) + h);
  int N = nt * 16 + (lane & 15);
  w1p[idx] = (__bf16)W1[((long)e * D_T + K) * H_T + N];
}

__global__ void moe_pack_w2_kernel(const float* __restrict__ W2, __bf16* __restrict__ w2p) {
  long idx = (long)blockIdx.x * 256 + threadIdx.x;
  if (idx >= (long)E_T * KT2 * NT2 * 512) return;
  int h    = (int)(idx & 15);
  int lane = (int)((idx >> 4) & 31);
  long rest = idx >> 9;
  int nt = (int)(rest % NT2); rest /= NT2;
  int kt = (int)(rest % KT2);
  int e  = (int)(rest / KT2);
  int K = kt * 32 + kperm(((lane >> 4) << 4) + h);
  int N = nt * 16 + (lane & 15);
  float v = (N < C_T) ? W2[((long)e * H_T + K) * C_T + N] : 0.0f;
  w2p[idx] = (__bf16)v;
}

// ---------------------------------------------------------------------------
// Router: logits[b][e] = dot(x[b], Wr[:,e]) + br[e].
// ---------------------------------------------------------------------------
__global__ __launch_bounds__(64) void moe_router_kernel(
    const float* __restrict__ x, const float* __restrict__ Wr,
    const float* __restrict__ br, float* __restrict__ logits) {
  const int b = blockIdx.x;
  const int e = threadIdx.x;
  __shared__ float sx[D_T];
  for (int i = e; i < D_T; i += 64) sx[i] = x[(long)b * D_T + i];
  __syncthreads();
  float acc = 0.0f;
  for (int d = 0; d < D_T; ++d) acc += sx[d] * Wr[d * E_T + e];
  logits[(long)b * E_T + e] = acc + br[e];
}

// ---------------------------------------------------------------------------
// Softmax + top-8 + routing lists + loss accumulators. One wave32 per token.
// ---------------------------------------------------------------------------
__global__ __launch_bounds__(128) void moe_topk_kernel(
    const float* __restrict__ logits, int* __restrict__ counts,
    int* __restrict__ toks, float* __restrict__ wts,
    float* __restrict__ colsum, float* __restrict__ zsum) {
  const int lane = threadIdx.x & 31;
  const int b = blockIdx.x * 4 + (threadIdx.x >> 5);
  const long base = (long)b * E_T;
  float l0 = logits[base + lane];
  float l1 = logits[base + lane + 32];

  float m = fmaxf(l0, l1);
  #pragma unroll
  for (int off = 16; off > 0; off >>= 1) m = fmaxf(m, __shfl_xor(m, off, 32));
  float p0 = __expf(l0 - m), p1 = __expf(l1 - m);
  float s = p0 + p1;
  #pragma unroll
  for (int off = 16; off > 0; off >>= 1) s += __shfl_xor(s, off, 32);
  const float lse = m + __logf(s);
  p0 /= s; p1 /= s;

  float kv[TOPK]; int ki[TOPK];
  #pragma unroll
  for (int it = 0; it < TOPK; ++it) {
    float v; int ix;
    if (p0 >= p1) { v = p0; ix = lane; } else { v = p1; ix = lane + 32; }
    #pragma unroll
    for (int off = 16; off > 0; off >>= 1) {
      float ov = __shfl_xor(v, off, 32);
      int   oi = __shfl_xor(ix, off, 32);
      if (ov > v || (ov == v && oi < ix)) { v = ov; ix = oi; }
    }
    kv[it] = v; ki[it] = ix;
    if (ix == lane) p0 = -1.0f;
    else if (ix == lane + 32) p1 = -1.0f;
  }

  if (lane == 0) {
    float ssum = 0.0f;
    #pragma unroll
    for (int j = 0; j < TOPK; ++j) ssum += kv[j];
    unsafeAtomicAdd(zsum, lse * lse);
    #pragma unroll
    for (int j = 0; j < TOPK; ++j) {
      float w = kv[j] / ssum;
      int ex = ki[j];
      unsafeAtomicAdd(&colsum[ex], w);
      int slot = atomicAdd(&counts[ex], 1);
      toks[ex * B_T + slot] = b;
      wts[ex * B_T + slot] = w;
    }
  }
}

// ---------------------------------------------------------------------------
// Sparse expert MLP: block = (expert e, 32-token tile), 8 waves, M=32.
// Layer1: relu(X @ W1[e] + b1[e]) -> 32x768 bf16 LDS tile (K-permuted).
//   Per wave: 2 M-tiles x 3 N-tiles per pass, 2 N passes. Each kt half-step:
//   5 fragment loads feed 6 WMMAs (B reused across both M tiles), ping-pong
//   double buffered, scheduler hinted to interleave WMMA with loads.
// Layer2: H @ W2[e] + b2[e]; gate-scaled atomic-add into final logits.
// ---------------------------------------------------------------------------
__global__ __launch_bounds__(256, 1) void moe_mlp_kernel(
    const __bf16* __restrict__ xbf, const __bf16* __restrict__ w1p,
    const __bf16* __restrict__ w2p, const float* __restrict__ b1,
    const float* __restrict__ b2, const int* __restrict__ counts,
    const int* __restrict__ toks, const float* __restrict__ wts,
    float* __restrict__ out) {
  const int e = blockIdx.x;
  const int tile = blockIdx.y;
  const int n_e = counts[e];
  if (tile * MT >= n_e) return;

  const int tid  = threadIdx.x;
  const int wave = tid >> 5;
  const int lane = tid & 31;
  const int row  = lane & 15;              // A row / D column within 16
  const int asel = (lane < 16) ? 0 : 16;   // A-fragment half-select (permuted)
  const int mbase = (lane < 16) ? 0 : 8;   // D-fragment row base

  __shared__ int   s_tok[MT];
  __shared__ float s_w[MT];
  __shared__ __align__(16) __bf16 s_h[MT * H_T];  // 48 KB hidden tile (K-permuted)

  if (tid < MT) {
    int idx = tile * MT + tid;
    if (idx < n_e) { s_tok[tid] = toks[e * B_T + idx]; s_w[tid] = wts[e * B_T + idx]; }
    else           { s_tok[tid] = 0;                   s_w[tid] = 0.0f; }
  }
  __syncthreads();

  const __bf16* aptr0 = xbf + (long)s_tok[row] * D_T + asel;       // M-tile 0
  const __bf16* aptr1 = xbf + (long)s_tok[16 + row] * D_T + asel;  // M-tile 1
  const long bstride1 = (long)NT1 * 512;  // per-kt stride in packed W1

  // ---- Layer 1: two N passes of (2 M-tiles x 3 N-tiles) per wave ----
  for (int pass = 0; pass < 2; ++pass) {
    const int ntbase = wave * 6 + pass * 3;
    const __bf16* w1b = w1p + ((long)e * KT1 * NT1 + ntbase) * 512 + lane * 16;

    v8f acc[2][3] = {};
    v16bf a0[2], a1[2], b0[3], b1v[3];
    a0[0] = load_v16(aptr0);
    a0[1] = load_v16(aptr1);
    #pragma unroll
    for (int j = 0; j < 3; ++j) b0[j] = load_v16(w1b + j * 512);

    for (int kt = 0; kt < KT1; kt += 2) {
      if (kt + 4 < KT1) __builtin_prefetch(w1b + (kt + 4) * bstride1, 0, 3);

      a1[0] = load_v16(aptr0 + (kt + 1) * 32);
      a1[1] = load_v16(aptr1 + (kt + 1) * 32);
      #pragma unroll
      for (int j = 0; j < 3; ++j) b1v[j] = load_v16(w1b + (kt + 1) * bstride1 + j * 512);
      #pragma unroll
      for (int j = 0; j < 3; ++j)
        #pragma unroll
        for (int m2 = 0; m2 < 2; ++m2)
          acc[m2][j] = __builtin_amdgcn_wmma_f32_16x16x32_bf16(
              false, a0[m2], false, b0[j], (short)0, acc[m2][j], false, false);
      moe_sched_hint();

      if (kt + 2 < KT1) {
        a0[0] = load_v16(aptr0 + (kt + 2) * 32);
        a0[1] = load_v16(aptr1 + (kt + 2) * 32);
        #pragma unroll
        for (int j = 0; j < 3; ++j) b0[j] = load_v16(w1b + (kt + 2) * bstride1 + j * 512);
      }
      #pragma unroll
      for (int j = 0; j < 3; ++j)
        #pragma unroll
        for (int m2 = 0; m2 < 2; ++m2)
          acc[m2][j] = __builtin_amdgcn_wmma_f32_16x16x32_bf16(
              false, a1[m2], false, b1v[j], (short)0, acc[m2][j], false, false);
      moe_sched_hint();
    }

    // bias + relu, store into LDS in the K-permuted A layout for layer 2
    #pragma unroll
    for (int j = 0; j < 3; ++j) {
      const int ncol = (ntbase + j) * 16 + row;   // hidden index = layer-2 K
      const float bias = b1[e * H_T + ncol];
      const int p = (ncol >> 5) * 32 + kperm(ncol & 31);
      #pragma unroll
      for (int m2 = 0; m2 < 2; ++m2)
        #pragma unroll
        for (int m = 0; m < 8; ++m) {
          float v = acc[m2][j][m] + bias;
          s_h[(m2 * 16 + mbase + m) * H_T + p] = (__bf16)(v > 0.0f ? v : 0.0f);
        }
    }
  }
  __syncthreads();

  // ---- Layer 2: waves 0-3 -> M-tile 0, waves 4-7 -> M-tile 1; 3 C-tiles each ----
  const int mtile   = wave >> 2;
  const int nt2base = (wave & 3) * 3;
  const __bf16* hptr = &s_h[(mtile * 16 + row) * H_T] + asel;
  const __bf16* w2b  = w2p + (long)e * KT2 * NT2 * 512 + nt2base * 512 + lane * 16;
  const long bstride2 = (long)NT2 * 512;

  v8f acc2[3] = {};
  {
    v16bf a0, a1, b0[3], b1v[3];
    a0 = load_v16(hptr);
    #pragma unroll
    for (int t = 0; t < 3; ++t) b0[t] = load_v16(w2b + t * 512);

    for (int kt = 0; kt < KT2; kt += 2) {
      a1 = load_v16(hptr + (kt + 1) * 32);
      #pragma unroll
      for (int t = 0; t < 3; ++t) b1v[t] = load_v16(w2b + (kt + 1) * bstride2 + t * 512);
      #pragma unroll
      for (int t = 0; t < 3; ++t)
        acc2[t] = __builtin_amdgcn_wmma_f32_16x16x32_bf16(
            false, a0, false, b0[t], (short)0, acc2[t], false, false);
      if (kt + 2 < KT2) {
        a0 = load_v16(hptr + (kt + 2) * 32);
        #pragma unroll
        for (int t = 0; t < 3; ++t) b0[t] = load_v16(w2b + (kt + 2) * bstride2 + t * 512);
      }
      #pragma unroll
      for (int t = 0; t < 3; ++t)
        acc2[t] = __builtin_amdgcn_wmma_f32_16x16x32_bf16(
            false, a1, false, b1v[t], (short)0, acc2[t], false, false);
    }
  }

  #pragma unroll
  for (int t = 0; t < 3; ++t) {
    const int c = (nt2base + t) * 16 + row;
    if (c < C_T) {
      const float bias = b2[e * C_T + c];
      #pragma unroll
      for (int m = 0; m < 8; ++m) {
        const int mm = mtile * 16 + mbase + m;
        const float w = s_w[mm];
        if (w != 0.0f)
          unsafeAtomicAdd(&out[(long)s_tok[mm] * C_T + c], (acc2[t][m] + bias) * w);
      }
    }
  }
}

// ---------------------------------------------------------------------------
// aux_loss = 0.1*mean(lse^2) + 0.1*sum((colmean - 1/E)^2)
// ---------------------------------------------------------------------------
__global__ __launch_bounds__(64) void moe_finalize_kernel(
    const float* __restrict__ colsum, const float* __restrict__ zsum,
    float* __restrict__ out_aux) {
  const int t = threadIdx.x;
  __shared__ float red[64];
  float c = colsum[t] / (float)B_T - 1.0f / (float)E_T;
  red[t] = c * c;
  __syncthreads();
  if (t == 0) {
    float s = 0.0f;
    for (int i = 0; i < 64; ++i) s += red[i];
    out_aux[0] = 0.1f * (zsum[0] / (float)B_T) + 0.1f * s;
  }
}

// ---------------------------------------------------------------------------
extern "C" void kernel_launch(void* const* d_in, const int* in_sizes, int n_in,
                              void* d_out, int out_size, void* d_ws, size_t ws_size,
                              hipStream_t stream) {
  (void)in_sizes; (void)n_in; (void)out_size; (void)ws_size;
  const float* x  = (const float*)d_in[0];   // [B, D]
  const float* Wr = (const float*)d_in[1];   // [D, E]
  const float* br = (const float*)d_in[2];   // [E]
  const float* W1 = (const float*)d_in[3];   // [E, D, H]
  const float* b1 = (const float*)d_in[4];   // [E, H]
  const float* W2 = (const float*)d_in[5];   // [E, H, C]
  const float* b2 = (const float*)d_in[6];   // [E, C]
  float* out = (float*)d_out;                // [B*C] final_logits ++ [1] aux

  char* ws = (char*)d_ws;
  size_t off = 0;
  auto take = [&](size_t bytes) -> char* {
    char* p = ws + off;
    off = (off + bytes + 255) & ~(size_t)255;
    return p;
  };
  float* logits = (float*)take((size_t)B_T * E_T * 4);
  int*   counts = (int*)  take((size_t)E_T * 4);
  float* colsum = (float*)take((size_t)E_T * 4);
  float* zsum   = (float*)take(256);
  int*   toks   = (int*)  take((size_t)E_T * B_T * 4);
  float* wts    = (float*)take((size_t)E_T * B_T * 4);
  __bf16* xbf   = (__bf16*)take((size_t)B_T * D_T * 2);
  __bf16* w1p   = (__bf16*)take((size_t)E_T * KT1 * NT1 * 512 * 2);
  __bf16* w2p   = (__bf16*)take((size_t)E_T * KT2 * NT2 * 512 * 2);

  // 1) zero outputs / accumulators
  {
    long n = (long)B_T * C_T;
    moe_zero_kernel<<<(unsigned)((n + 255) / 256), 256, 0, stream>>>(out, counts, colsum, zsum);
  }
  // 2) pack activations & weights to bf16 WMMA layouts
  {
    long n = (long)B_T * D_T;
    moe_pack_x_kernel<<<(unsigned)((n + 255) / 256), 256, 0, stream>>>(x, xbf);
  }
  {
    long n = (long)E_T * KT1 * NT1 * 512;
    moe_pack_w1_kernel<<<(unsigned)((n + 255) / 256), 256, 0, stream>>>(W1, w1p);
  }
  {
    long n = (long)E_T * KT2 * NT2 * 512;
    moe_pack_w2_kernel<<<(unsigned)((n + 255) / 256), 256, 0, stream>>>(W2, w2p);
  }
  // 3) router + routing/losses
  moe_router_kernel<<<B_T, 64, 0, stream>>>(x, Wr, br, logits);
  moe_topk_kernel<<<B_T / 4, 128, 0, stream>>>(logits, counts, toks, wts, colsum, zsum);
  // 4) sparse expert MLP (WMMA bf16, M=32, double-buffered + sched hints)
  moe_mlp_kernel<<<dim3(E_T, B_T / MT), 256, 0, stream>>>(xbf, w1p, w2p, b1, b2,
                                                          counts, toks, wts, out);
  // 5) aux loss
  moe_finalize_kernel<<<1, 64, 0, stream>>>(colsum, zsum, out + (size_t)B_T * C_T);
}
